// TSTabFusionMLM_32220844654738
// MI455X (gfx1250) — matrix operand order repaired and checked by reading, not compiled, not run on hardware
//
#include <hip/hip_runtime.h>
#include <hip/hip_bf16.h>
#include <stdint.h>

// ---------------------------------------------------------------------------
// TSTabFusionMLM forward for MI455X (gfx1250, wave32, WMMA).
// All large matmuls go through one generic f16 WMMA GEMM kernel
// (v_wmma_f32_16x16x32_f16), fp32 accumulate, fp32 residual/LN stream.
// Tiling: 256 threads = 8 waves; block tile 32 x (64*NT); each wave computes
// NT 16x16 tiles, reusing its A fragment across NT WMMA issues.
// A tile is staged with gfx1250 async global->LDS copies when the toolchain
// exposes the builtins (ASYNCcnt protocol), else with 8B vector copies.
// ---------------------------------------------------------------------------

typedef __attribute__((ext_vector_type(16))) _Float16 v16h;
typedef __attribute__((ext_vector_type(8)))  float    v8f;
typedef __attribute__((ext_vector_type(4)))  float    v4f;
typedef __attribute__((ext_vector_type(4)))  _Float16 v4h;
typedef __attribute__((ext_vector_type(2)))  int      v2i;

#if defined(__has_builtin)
#if __has_builtin(__builtin_amdgcn_global_load_async_to_lds_b64) && \
    __has_builtin(__builtin_amdgcn_s_wait_asynccnt)
#define USE_ASYNC_LDS 1
#endif
#endif
#ifndef USE_ASYNC_LDS
#define USE_ASYNC_LDS 0
#endif

#if USE_ASYNC_LDS
typedef __attribute__((address_space(1))) v2i* gv2i_p;   // global 64-bit payload
typedef __attribute__((address_space(3))) v2i* lv2i_p;   // LDS 64-bit payload
#endif

#define BSZ    32
#define CIN    32
#define SEQL   512
#define DM     256
#define NCATN  8
#define NCONTN 16
#define VOC    100
#define NHEAD  8
#define NLAY   3
#define LTOK   536        // SEQL + NCATN + NCONTN
#define LPAD   544        // padded to multiple of 32 for WMMA K / 16 for M
#define MTOK   17152      // BSZ*LTOK
#define MTS    16384      // BSZ*SEQL

// ------------------------------ utilities ----------------------------------

__device__ __forceinline__ uint32_t hashu(uint32_t x) {
  x ^= x >> 16; x *= 0x7feb352dU;
  x ^= x >> 15; x *= 0x846ca68bU;
  x ^= x >> 16; return x;
}
__device__ __forceinline__ float u01(uint32_t h) {
  return (float)(h >> 8) * (1.0f / 16777216.0f);
}
__device__ __forceinline__ float gelu_exact(float v) {
  return 0.5f * v * (1.0f + erff(v * 0.70710678118654752f));
}

// --------------------------- generic WMMA GEMM -----------------------------
// C[z] = A[z] (MxK, row major, f16) * B[z] (KxN, row major, f16) + bias
// NT = 16-col tiles per wave.  Block tile = 32 rows x 64*NT cols.
// Requires: M % 32 == 0, K % 32 == 0.  N arbitrary (guarded).
template <int NT>
__global__ __launch_bounds__(256)
void gemm_wmma(const _Float16* __restrict__ A, long long sA,
               const _Float16* __restrict__ B, long long sB,
               const float* __restrict__ bias, long long sBias,
               float* __restrict__ Cf, _Float16* __restrict__ Ch, long long sC,
               int M, int N, int K)
{
  constexpr int TN = 64 * NT;                 // block tile columns
  constexpr int CG = TN / 4;                  // 4-col groups per tile
  constexpr int RSTEP = 256 / CG;             // row step in B staging
  __shared__ __align__(16) _Float16 lA[32 * 32];     // A tile, row major
  __shared__ __align__(16) _Float16 lBt[TN * 32];    // B tile, transposed [col][k]
  const int z = blockIdx.z;
  A += (long long)z * sA;
  B += (long long)z * sB;
  const int t    = threadIdx.x;
  const int wave = t >> 5;          // 0..7
  const int lane = t & 31;
  const int lm   = lane & 15;
  const int hi   = lane >> 4;       // half-wave select
  const int wr   = wave >> 2;       // wave row group (0/1)
  const int wc   = wave & 3;        // wave col group (0..3)
  const int rowBase = blockIdx.y << 5;
  const int colBase = blockIdx.x * TN;
  v8f acc[NT] = {};

  // B staging geometry (vectorized 4-wide along N; one uniform guard / group)
  const int colv  = (t % CG) * 4;             // col within tile (mult of 4)
  const int rbase = t / CG;                   // starting row
  const int gcolB = colBase + colv;
  const bool vecOK = (gcolB + 4 <= N);

  for (int k0 = 0; k0 < K; k0 += 32) {
    __syncthreads();
    { // stage A: 1024 halves, 8 bytes per thread
      const int idx = t << 2;
      const int r = idx >> 5, c = idx & 31;
      const _Float16* gsrc = &A[(long long)(rowBase + r) * K + (k0 + c)];
#if USE_ASYNC_LDS
      __builtin_amdgcn_global_load_async_to_lds_b64(
          (gv2i_p)(void*)gsrc, (lv2i_p)&lA[idx], 0, 0);
#else
      *(v4h*)&lA[idx] = *(const v4h*)gsrc;
#endif
    }
    { // stage B transposed: lBt[c*32 + r] = B[k0+r][colBase+c]
      #pragma unroll
      for (int ii = 0; ii < 32 / RSTEP; ++ii) {
        const int r = rbase + ii * RSTEP;
        if (vecOK) {
          const v4h bv = *(const v4h*)&B[(long long)(k0 + r) * N + gcolB];
          #pragma unroll
          for (int u = 0; u < 4; ++u) lBt[(colv + u) * 32 + r] = bv[u];
        } else {
          #pragma unroll
          for (int u = 0; u < 4; ++u) {
            const int gc = gcolB + u;
            lBt[(colv + u) * 32 + r] =
                (gc < N) ? B[(long long)(k0 + r) * N + gc] : (_Float16)0.f;
          }
        }
      }
    }
    if (k0 + 32 < K) { // prefetch next A tile -> global_prefetch_b8
      __builtin_prefetch(&A[(long long)(rowBase + (t >> 3)) * K + (k0 + 32)], 0, 1);
    }
#if USE_ASYNC_LDS
    // Async LDS writes are tracked by ASYNCcnt (not DScnt): drain own async
    // ops before signaling the barrier so post-barrier reads are safe.
    __builtin_amdgcn_s_wait_asynccnt(0);
#endif
    __syncthreads();

    // Fragments per CDNA5 16-bit layouts (CDNA5 ISA 7.12.2).
    union F16 { v16h v; v4f q[2]; };
    F16 fa;
    const int aRow = (wr << 4) + lm;
    const int kbA  = hi << 3;                        // lanes 16-31 start at K=8
    fa.q[0] = *(const v4f*)&lA[(aRow << 5) + kbA];       // K = kbA .. kbA+7
    fa.q[1] = *(const v4f*)&lA[(aRow << 5) + kbA + 16];  // K = kbA+16 .. kbA+23
    const int kbB = hi << 3;                         // lanes 16-31 start at K=8
    #pragma unroll
    for (int j = 0; j < NT; ++j) {
      F16 fb;
      const _Float16* bp = &lBt[((wc * NT + j) * 16 + lm) << 5];
      fb.q[0] = *(const v4f*)&bp[kbB];
      fb.q[1] = *(const v4f*)&bp[kbB + 16];
      acc[j] = __builtin_amdgcn_wmma_f32_16x16x32_f16(
          false, fa.v, false, fb.v, (short)0, acc[j], false, false);
    }
  }

  #pragma unroll
  for (int j = 0; j < NT; ++j) {
    const int gcol = colBase + (wc * NT + j) * 16 + lm;
    if (gcol < N) {
      const float bv = bias ? bias[(long long)z * sBias + gcol] : 0.0f;
      #pragma unroll
      for (int r = 0; r < 8; ++r) {
        const long long o = (long long)z * sC +
            (long long)(rowBase + (wr << 4) + (hi << 3) + r) * N + gcol;
        const float vv = acc[j][r] + bv;
        if (Cf) Cf[o] = vv;
        if (Ch) Ch[o] = (_Float16)vv;
      }
    }
  }
}

// --------------------------- conversion kernels ----------------------------

__global__ void cvtF16(const float* __restrict__ s, _Float16* __restrict__ d,
                       long long n) {
  long long i = (long long)blockIdx.x * 256 + threadIdx.x;
  if (i < n) d[i] = (_Float16)s[i];
}

// WP_w (D, CIN) -> transposed f16 (CIN, D)
__global__ void cvt_wp(const float* __restrict__ W, _Float16* __restrict__ Bt) {
  int i = blockIdx.x * 256 + threadIdx.x;      // i over 32*256
  int c = i >> 8, d = i & 255;
  Bt[i] = (_Float16)W[(long long)d * CIN + c];
}

// --------------------------- masking kernels -------------------------------

// Builds the (MTS x 32) f16 A-matrix for the input projection with masked ts.
__global__ void mask_ts_pack(const float* __restrict__ xts,
                             _Float16* __restrict__ A) {
  int i = blockIdx.x * 256 + threadIdx.x;      // (b,t), 16384 total
  if (i >= MTS) return;
  int b = i >> 9, t = i & 511;
  uint32_t h0 = hashu(0x9E3779B9u ^ (uint32_t)i);
  uint32_t h1 = hashu(0x85EBCA6Bu ^ (uint32_t)i);
  uint32_t h2 = hashu(0xC2B2AE35u ^ (uint32_t)i);
  float u = u01(h0), rv = u01(h1);
  int srcT = t; float mul = 1.0f;
  if (u < 0.15f) {
    if (rv < 0.8f)       mul = 0.0f;
    else if (rv < 0.9f)  srcT = (int)(h2 & 511u);
  }
  const float* base = xts + (long long)b * CIN * SEQL;
  #pragma unroll 4
  for (int c = 0; c < CIN; ++c)
    A[(long long)i * CIN + c] = (_Float16)(mul * base[c * SEQL + srcT]);
}

__global__ void mask_cat_k(const int* __restrict__ xcat, int* __restrict__ mc) {
  int i = threadIdx.x;                         // 256 = BSZ*NCATN
  if (i >= BSZ * NCATN) return;
  uint32_t h0 = hashu(0x27220A95u ^ (uint32_t)i);
  uint32_t h1 = hashu(0x165667B1u ^ (uint32_t)i);
  uint32_t h2 = hashu(0xFD7046C5u ^ (uint32_t)i);
  float u = u01(h0), rv = u01(h1);
  int v = xcat[i];
  if (u < 0.15f) {
    if (rv < 0.8f)       v = 0;
    else if (rv < 0.9f)  v = (int)(h2 % (uint32_t)VOC);
  }
  mc[i] = v;
}

__global__ void mask_cont_k(const float* __restrict__ xc, float* __restrict__ mc) {
  __shared__ float mean[NCONTN];
  int t = threadIdx.x;                         // 512 threads
  if (t < NCONTN) {
    float s = 0.f;
    for (int b = 0; b < BSZ; ++b) s += xc[b * NCONTN + t];
    mean[t] = s / (float)BSZ;
  }
  __syncthreads();
  if (t < BSZ * NCONTN) {
    int q = t & (NCONTN - 1);
    uint32_t h0 = hashu(0xB5297A4Du ^ (uint32_t)t);
    uint32_t h1 = hashu(0x68E31DA4u ^ (uint32_t)t);
    uint32_t h2 = hashu(0x1B56C4E9u ^ (uint32_t)t);
    float u = u01(h0), rv = u01(h1);
    float v = xc[t];
    if (u < 0.15f) {
      if (rv < 0.8f)       v = mean[q];
      else if (rv < 0.9f)  v = xc[(h2 & 31u) * NCONTN + q];
    }
    mc[t] = v;
  }
}

// ----------------------- encoder input assembly ----------------------------

__global__ void assemble_x(const float* __restrict__ proj,
                           const int* __restrict__ mcat,
                           const float* __restrict__ mcont,
                           const float* __restrict__ emb,
                           const float* __restrict__ contw,
                           const float* __restrict__ contb,
                           const float* __restrict__ pos,
                           float* __restrict__ x, _Float16* __restrict__ xh) {
  const int row = blockIdx.x;                  // 17152 rows
  const int b = row / LTOK, l = row % LTOK, d = threadIdx.x;
  float v;
  if (l < SEQL) {
    v = proj[((long long)b * SEQL + l) * DM + d];
  } else if (l < SEQL + NCATN) {
    int n = l - SEQL;
    int idx = mcat[b * NCATN + n];
    idx = idx < 0 ? 0 : (idx >= VOC ? VOC - 1 : idx);
    v = emb[((long long)n * VOC + idx) * DM + d];
  } else {
    int q = l - SEQL - NCATN;
    v = mcont[b * NCONTN + q] * contw[d] + contb[d];
  }
  v += pos[(long long)l * DM + d];
  const long long o = (long long)row * DM + d;
  x[o] = v; xh[o] = (_Float16)v;
}

// ------------------------- attention pack/unpack ---------------------------

// tmp16 (MTOK x 256) -> padded (B*H, LPAD, 32) [normal] or (B*H, 32, LPAD) [T]
__global__ void pack_qkv(const _Float16* __restrict__ t16,
                         _Float16* __restrict__ dst, int transposed) {
  long long i = (long long)blockIdx.x * 256 + threadIdx.x;
  if (i >= (long long)BSZ * NHEAD * LPAD * 32) return;
  int d = (int)(i & 31);
  long long r = i >> 5;
  int l = (int)(r % LPAD);
  long long bh = r / LPAD;
  int b = (int)(bh >> 3), h = (int)(bh & 7);
  _Float16 v = (_Float16)0.f;
  if (l < LTOK) v = t16[((long long)b * LTOK + l) * DM + h * 32 + d];
  if (!transposed) dst[i] = v;
  else             dst[(bh * 32 + d) * LPAD + l] = v;
}

__global__ void unpack_o(const _Float16* __restrict__ Op,
                         _Float16* __restrict__ oA) {
  long long i = (long long)blockIdx.x * 256 + threadIdx.x;
  if (i >= (long long)MTOK * DM) return;
  int dd = (int)(i & 255);
  long long row = i >> 8;
  int b = (int)(row / LTOK), l = (int)(row % LTOK);
  int h = dd >> 5, d = dd & 31;
  oA[i] = Op[(((long long)b * NHEAD + h) * LPAD + l) * 32 + d];
}

// Softmax over valid columns; scale folded in; pads written to zero; in-place.
__global__ void softmax_rows(_Float16* __restrict__ SP, int Lvalid, float scale) {
  _Float16* row = SP + ((long long)blockIdx.y * LPAD + blockIdx.x) * LPAD;
  __shared__ float red[256];
  const int t = threadIdx.x;
  float mx = -1e30f;
  for (int j = t; j < Lvalid; j += 256) mx = fmaxf(mx, (float)row[j] * scale);
  red[t] = mx; __syncthreads();
  for (int s = 128; s > 0; s >>= 1) { if (t < s) red[t] = fmaxf(red[t], red[t + s]); __syncthreads(); }
  mx = red[0]; __syncthreads();
  float sum = 0.f;
  for (int j = t; j < Lvalid; j += 256) {
    float v = __expf((float)row[j] * scale - mx);
    sum += v; row[j] = (_Float16)v;
  }
  red[t] = sum; __syncthreads();
  for (int s = 128; s > 0; s >>= 1) { if (t < s) red[t] += red[t + s]; __syncthreads(); }
  const float inv = 1.0f / red[0];
  __syncthreads();
  for (int j = t; j < LPAD; j += 256) {
    float v = (j < Lvalid) ? (float)row[j] * inv : 0.0f;
    row[j] = (_Float16)v;
  }
}

// ------------------------- LN / GELU kernels -------------------------------

// x = LN(x + y); writes f32 x (in place) and f16 mirror.  Width = 256.
__global__ void add_ln256(float* __restrict__ x, const float* __restrict__ y,
                          const float* __restrict__ s, const float* __restrict__ b,
                          _Float16* __restrict__ xh) {
  const long long row = blockIdx.x;
  const int t = threadIdx.x;
  const long long o = row * DM + t;
  float v = x[o] + (y ? y[o] : 0.0f);
  __shared__ float red[256];
  red[t] = v; __syncthreads();
  for (int st = 128; st > 0; st >>= 1) { if (t < st) red[t] += red[t + st]; __syncthreads(); }
  const float m = red[0] * (1.0f / DM); __syncthreads();
  const float dv = v - m;
  red[t] = dv * dv; __syncthreads();
  for (int st = 128; st > 0; st >>= 1) { if (t < st) red[t] += red[t + st]; __syncthreads(); }
  const float inv = rsqrtf(red[0] * (1.0f / DM) + 1e-5f);
  const float ov = dv * inv * s[t] + b[t];
  x[o] = ov; xh[o] = (_Float16)ov;
}

// In-place LN over width 512 on an f16 buffer (ts head).
__global__ void ln512_f16(_Float16* __restrict__ h, const float* __restrict__ s,
                          const float* __restrict__ b) {
  const long long row = blockIdx.x;
  const int t = threadIdx.x;
  _Float16* r = h + row * 512;
  float v0 = (float)r[t], v1 = (float)r[t + 256];
  __shared__ float red[256];
  red[t] = v0 + v1; __syncthreads();
  for (int st = 128; st > 0; st >>= 1) { if (t < st) red[t] += red[t + st]; __syncthreads(); }
  const float m = red[0] * (1.0f / 512.0f); __syncthreads();
  const float d0 = v0 - m, d1 = v1 - m;
  red[t] = d0 * d0 + d1 * d1; __syncthreads();
  for (int st = 128; st > 0; st >>= 1) { if (t < st) red[t] += red[t + st]; __syncthreads(); }
  const float inv = rsqrtf(red[0] * (1.0f / 512.0f) + 1e-5f);
  r[t]       = (_Float16)(d0 * inv * s[t] + b[t]);
  r[t + 256] = (_Float16)(d1 * inv * s[t + 256] + b[t + 256]);
}

__global__ void gelu16(_Float16* __restrict__ h, long long n) {
  long long i = (long long)blockIdx.x * 256 + threadIdx.x;
  if (i < n) h[i] = (_Float16)gelu_exact((float)h[i]);
}

// ---------------------------- head extraction ------------------------------

__global__ void extract_ts(const _Float16* __restrict__ xh, _Float16* __restrict__ A) {
  long long i = (long long)blockIdx.x * 256 + threadIdx.x;  // MTS*256
  long long row = i >> 8; int d = (int)(i & 255);
  long long b = row >> 9; int t = (int)(row & 511);
  A[i] = xh[((long long)b * LTOK + t) * DM + d];
}
__global__ void extract_cat(const _Float16* __restrict__ xh, _Float16* __restrict__ A) {
  int i = blockIdx.x * 256 + threadIdx.x;                   // 8*32*256
  int d = i & 255; int r = i >> 8; int n = r >> 5; int b = r & 31;
  A[i] = xh[((long long)b * LTOK + SEQL + n) * DM + d];
}
__global__ void extract_cont(const _Float16* __restrict__ xh, _Float16* __restrict__ A) {
  int i = blockIdx.x * 256 + threadIdx.x;                   // 512*256
  int d = i & 255; int r = i >> 8; int b = r >> 4; int q = r & 15;
  A[i] = xh[((long long)b * LTOK + SEQL + NCATN + q) * DM + d];
}

// ---------------------------- output scatter -------------------------------

__global__ void out_ts(const float* __restrict__ yts, float* __restrict__ out) {
  int i = blockIdx.x * 256 + threadIdx.x;                   // BSZ*CIN*SEQL
  if (i >= BSZ * CIN * SEQL) return;
  int b = i / (CIN * SEQL); int rem = i - b * CIN * SEQL;
  int c = rem / SEQL; int t = rem - c * SEQL;
  out[i] = yts[((long long)b * SEQL + t) * CIN + c];
}
__global__ void out_cat(const float* __restrict__ yc, float* __restrict__ out) {
  int i = blockIdx.x * 256 + threadIdx.x;                   // 32*8*100
  if (i >= BSZ * NCATN * VOC) return;
  int b = i / (NCATN * VOC); int rem = i - b * NCATN * VOC;
  int n = rem / VOC; int k = rem - n * VOC;
  out[i] = yc[((long long)n * BSZ + b) * VOC + k];
}
__global__ void out_cont(const float* __restrict__ yq, float* __restrict__ out) {
  int i = blockIdx.x * 256 + threadIdx.x;                   // 512
  if (i < BSZ * NCONTN) out[i] = yq[i];
}

// ------------------------------- launcher ----------------------------------

extern "C" void kernel_launch(void* const* d_in, const int* in_sizes, int n_in,
                              void* d_out, int out_size, void* d_ws, size_t ws_size,
                              hipStream_t stream) {
  (void)in_sizes; (void)out_size; (void)ws_size;
  if (n_in < 71) return;

  // --- inputs (setup_inputs order; params pytree flattened in sorted-key order)
  const float* x_ts   = (const float*)d_in[0];
  const int*   x_cat  = (const int*)  d_in[1];
  const float* x_cont = (const float*)d_in[2];
  const float* WP_b    = (const float*)d_in[3];
  const float* WP_w    = (const float*)d_in[4];
  const float* cat_W1  = (const float*)d_in[5];
  const float* cat_W2  = (const float*)d_in[6];
  const float* cat_b1  = (const float*)d_in[7];
  const float* cat_b2  = (const float*)d_in[8];
  const float* cont_W1 = (const float*)d_in[9];
  const float* cont_W2 = (const float*)d_in[10];
  const float* cont_b  = (const float*)d_in[11];
  const float* cont_b1 = (const float*)d_in[12];
  const float* cont_b2 = (const float*)d_in[13];
  const float* cont_w  = (const float*)d_in[14];
  const float* emb     = (const float*)d_in[15];
  const int lb = 16;  // layers: per layer W1,W2,Wk,Wo,Wq,Wv,b1,b2,bk,bo,bq,bv,ln1_b,ln1_s,ln2_b,ln2_s
  const float* pos     = (const float*)d_in[lb + 48];
  const float* ts_W1   = (const float*)d_in[lb + 49];
  const float* ts_W2   = (const float*)d_in[lb + 50];
  const float* ts_b1   = (const float*)d_in[lb + 51];
  const float* ts_b2   = (const float*)d_in[lb + 52];
  const float* ts_ln_b = (const float*)d_in[lb + 53];
  const float* ts_ln_s = (const float*)d_in[lb + 54];
  float* out = (float*)d_out;

  // --- workspace bump allocator
  char* wsb = (char*)d_ws; size_t off = 0;
  auto alloc = [&](size_t bytes) -> void* {
    off = (off + 255) & ~(size_t)255;
    void* p = wsb + off; off += bytes; return p;
  };
  auto cvt = [&](const float* s, _Float16* d, long long n) {
    cvtF16<<<(int)((n + 255) / 256), 256, 0, stream>>>(s, d, n);
  };
  auto gemm = [&](const _Float16* A, long long sA, const _Float16* B, long long sB,
                  const float* bias, long long sBias,
                  float* Cf, _Float16* Ch, long long sC,
                  int M, int N, int K, int batch) {
    if (N % 128 == 0) {
      dim3 g(N / 128, M / 32, batch);
      gemm_wmma<2><<<g, 256, 0, stream>>>(A, sA, B, sB, bias, sBias, Cf, Ch, sC, M, N, K);
    } else {
      dim3 g((N + 63) / 64, M / 32, batch);
      gemm_wmma<1><<<g, 256, 0, stream>>>(A, sA, B, sB, bias, sBias, Cf, Ch, sC, M, N, K);
    }
  };

  // --- f16 weights
  _Float16* WPtH   = (_Float16*)alloc((size_t)CIN * DM * 2);
  _Float16* WqH[NLAY], *WkH[NLAY], *WvH[NLAY], *WoH[NLAY], *W1H[NLAY], *W2H[NLAY];
  for (int i = 0; i < NLAY; ++i) {
    W1H[i] = (_Float16*)alloc((size_t)DM * 4 * DM * 2);
    W2H[i] = (_Float16*)alloc((size_t)4 * DM * DM * 2);
    WkH[i] = (_Float16*)alloc((size_t)DM * DM * 2);
    WoH[i] = (_Float16*)alloc((size_t)DM * DM * 2);
    WqH[i] = (_Float16*)alloc((size_t)DM * DM * 2);
    WvH[i] = (_Float16*)alloc((size_t)DM * DM * 2);
  }
  _Float16* tsW1H  = (_Float16*)alloc((size_t)DM * 512 * 2);
  _Float16* tsW2H  = (_Float16*)alloc((size_t)512 * CIN * 2);
  _Float16* catW1H = (_Float16*)alloc((size_t)NCATN * DM * DM * 2);
  _Float16* catW2H = (_Float16*)alloc((size_t)NCATN * DM * VOC * 2);
  _Float16* cW1H   = (_Float16*)alloc((size_t)DM * DM * 2);
  _Float16* cW2H   = (_Float16*)alloc((size_t)DM * 2);

  // --- activations / scratch
  _Float16* Aproj  = (_Float16*)alloc((size_t)MTS * CIN * 2);
  float*    projOut= (float*)   alloc((size_t)MTS * DM * 4);
  int*      mcat   = (int*)     alloc((size_t)BSZ * NCATN * 4);
  float*    mcont  = (float*)   alloc((size_t)BSZ * NCONTN * 4);
  float*    xf     = (float*)   alloc((size_t)MTOK * DM * 4);
  _Float16* xh     = (_Float16*)alloc((size_t)MTOK * DM * 2);
  _Float16* tmp16  = (_Float16*)alloc((size_t)MTOK * DM * 2);
  _Float16* Qp     = (_Float16*)alloc((size_t)BSZ * NHEAD * LPAD * 32 * 2);
  _Float16* Ktp    = (_Float16*)alloc((size_t)BSZ * NHEAD * LPAD * 32 * 2);
  _Float16* Vp     = (_Float16*)alloc((size_t)BSZ * NHEAD * LPAD * 32 * 2);
  const int BHCHUNK = 64;  // attention batch-head chunk (4 chunks of 64)
  _Float16* SP     = (_Float16*)alloc((size_t)BHCHUNK * LPAD * LPAD * 2);
  _Float16* Op     = (_Float16*)alloc((size_t)BSZ * NHEAD * LPAD * 32 * 2);
  _Float16* oA     = (_Float16*)alloc((size_t)MTOK * DM * 2);
  float*    y2     = (float*)   alloc((size_t)MTOK * DM * 4);
  _Float16* h16    = (_Float16*)alloc((size_t)MTOK * 4 * DM * 2);
  _Float16* Ats    = (_Float16*)alloc((size_t)MTS * DM * 2);
  _Float16* hts    = (_Float16*)alloc((size_t)MTS * 512 * 2);
  float*    ytsOut = (float*)   alloc((size_t)MTS * CIN * 4);
  _Float16* Acat   = (_Float16*)alloc((size_t)NCATN * BSZ * DM * 2);
  _Float16* hcat   = (_Float16*)alloc((size_t)NCATN * BSZ * DM * 2);
  float*    ycat2  = (float*)   alloc((size_t)NCATN * BSZ * VOC * 4);
  _Float16* Acont  = (_Float16*)alloc((size_t)BSZ * NCONTN * DM * 2);
  _Float16* hcont  = (_Float16*)alloc((size_t)BSZ * NCONTN * DM * 2);
  float*    ycont2 = (float*)   alloc((size_t)BSZ * NCONTN * 4);

  // --- weight conversion
  cvt_wp<<<(CIN * DM) / 256, 256, 0, stream>>>(WP_w, WPtH);
  for (int i = 0; i < NLAY; ++i) {
    const int b0 = lb + i * 16;
    cvt((const float*)d_in[b0 + 0], W1H[i], (long long)DM * 4 * DM);
    cvt((const float*)d_in[b0 + 1], W2H[i], (long long)4 * DM * DM);
    cvt((const float*)d_in[b0 + 2], WkH[i], (long long)DM * DM);
    cvt((const float*)d_in[b0 + 3], WoH[i], (long long)DM * DM);
    cvt((const float*)d_in[b0 + 4], WqH[i], (long long)DM * DM);
    cvt((const float*)d_in[b0 + 5], WvH[i], (long long)DM * DM);
  }
  cvt(ts_W1, tsW1H, (long long)DM * 512);
  cvt(ts_W2, tsW2H, (long long)512 * CIN);
  cvt(cat_W1, catW1H, (long long)NCATN * DM * DM);
  cvt(cat_W2, catW2H, (long long)NCATN * DM * VOC);
  cvt(cont_W1, cW1H, (long long)DM * DM);
  cvt(cont_W2, cW2H, (long long)DM);

  // --- masking + input projection + assembly
  mask_ts_pack<<<(MTS + 255) / 256, 256, 0, stream>>>(x_ts, Aproj);
  mask_cat_k<<<1, 256, 0, stream>>>(x_cat, mcat);
  mask_cont_k<<<1, 512, 0, stream>>>(x_cont, mcont);
  gemm(Aproj, 0, WPtH, 0, WP_b, 0, projOut, nullptr, 0, MTS, DM, CIN, 1);
  assemble_x<<<MTOK, 256, 0, stream>>>(projOut, mcat, mcont, emb, cont_w, cont_b,
                                       pos, xf, xh);

  // --- transformer layers
  const long long sQ  = (long long)LPAD * 32;
  const long long sS  = (long long)LPAD * LPAD;
  const long long nPk = (long long)BSZ * NHEAD * LPAD * 32;
  for (int i = 0; i < NLAY; ++i) {
    const int b0 = lb + i * 16;
    const float* b1    = (const float*)d_in[b0 + 6];
    const float* b2    = (const float*)d_in[b0 + 7];
    const float* bk    = (const float*)d_in[b0 + 8];
    const float* bo    = (const float*)d_in[b0 + 9];
    const float* bq    = (const float*)d_in[b0 + 10];
    const float* bv    = (const float*)d_in[b0 + 11];
    const float* ln1_b = (const float*)d_in[b0 + 12];
    const float* ln1_s = (const float*)d_in[b0 + 13];
    const float* ln2_b = (const float*)d_in[b0 + 14];
    const float* ln2_s = (const float*)d_in[b0 + 15];

    // QKV projections -> packed per-(b,h) tiles (K transposed)
    gemm(xh, 0, WqH[i], 0, bq, 0, nullptr, tmp16, 0, MTOK, DM, DM, 1);
    pack_qkv<<<(int)((nPk + 255) / 256), 256, 0, stream>>>(tmp16, Qp, 0);
    gemm(xh, 0, WkH[i], 0, bk, 0, nullptr, tmp16, 0, MTOK, DM, DM, 1);
    pack_qkv<<<(int)((nPk + 255) / 256), 256, 0, stream>>>(tmp16, Ktp, 1);
    gemm(xh, 0, WvH[i], 0, bv, 0, nullptr, tmp16, 0, MTOK, DM, DM, 1);
    pack_qkv<<<(int)((nPk + 255) / 256), 256, 0, stream>>>(tmp16, Vp, 0);

    // attention in chunks of 64 batch-heads
    for (int ch = 0; ch < (BSZ * NHEAD) / BHCHUNK; ++ch) {
      const long long zo = (long long)ch * BHCHUNK;
      gemm(Qp + zo * sQ, sQ, Ktp + zo * sQ, sQ, nullptr, 0,
           nullptr, SP, sS, LPAD, LPAD, 32, BHCHUNK);
      softmax_rows<<<dim3(LPAD, BHCHUNK), 256, 0, stream>>>(SP, LTOK,
                                                            0.17677669529663689f);
      gemm(SP, sS, Vp + zo * sQ, sQ, nullptr, 0,
           nullptr, Op + zo * sQ, sQ, LPAD, 32, LPAD, BHCHUNK);
    }
    unpack_o<<<MTOK, 256, 0, stream>>>(Op, oA);
    gemm(oA, 0, WoH[i], 0, bo, 0, y2, nullptr, 0, MTOK, DM, DM, 1);
    add_ln256<<<MTOK, 256, 0, stream>>>(xf, y2, ln1_s, ln1_b, xh);

    // FFN
    gemm(xh, 0, W1H[i], 0, b1, 0, nullptr, h16, 0, MTOK, 4 * DM, DM, 1);
    gelu16<<<(int)(((long long)MTOK * 4 * DM + 255) / 256), 256, 0, stream>>>(
        h16, (long long)MTOK * 4 * DM);
    gemm(h16, 0, W2H[i], 0, b2, 0, y2, nullptr, 0, MTOK, DM, 4 * DM, 1);
    add_ln256<<<MTOK, 256, 0, stream>>>(xf, y2, ln2_s, ln2_b, xh);
  }

  // --- heads
  extract_ts<<<MTS, 256, 0, stream>>>(xh, Ats);
  extract_cat<<<(NCATN * BSZ * DM) / 256, 256, 0, stream>>>(xh, Acat);
  extract_cont<<<(BSZ * NCONTN * DM) / 256, 256, 0, stream>>>(xh, Acont);

  // ts head: gelu -> LN512 -> W2 -> transpose into d_out
  gemm(Ats, 0, tsW1H, 0, ts_b1, 0, nullptr, hts, 0, MTS, 512, DM, 1);
  gelu16<<<(int)(((long long)MTS * 512 + 255) / 256), 256, 0, stream>>>(
      hts, (long long)MTS * 512);
  ln512_f16<<<MTS, 256, 0, stream>>>(hts, ts_ln_s, ts_ln_b);
  gemm(hts, 0, tsW2H, 0, ts_b2, 0, ytsOut, nullptr, 0, MTS, CIN, 512, 1);
  out_ts<<<(BSZ * CIN * SEQL + 255) / 256, 256, 0, stream>>>(ytsOut, out);

  // cat head: batched over NCAT
  gemm(Acat, (long long)BSZ * DM, catW1H, (long long)DM * DM, cat_b1, DM,
       nullptr, hcat, (long long)BSZ * DM, BSZ, DM, DM, NCATN);
  gelu16<<<(NCATN * BSZ * DM + 255) / 256, 256, 0, stream>>>(
      hcat, (long long)NCATN * BSZ * DM);
  gemm(hcat, (long long)BSZ * DM, catW2H, (long long)DM * VOC, cat_b2, VOC,
       ycat2, nullptr, (long long)BSZ * VOC, BSZ, VOC, DM, NCATN);
  out_cat<<<(BSZ * NCATN * VOC + 255) / 256, 256, 0, stream>>>(
      ycat2, out + (long long)BSZ * CIN * SEQL);

  // cont head
  gemm(Acont, 0, cW1H, 0, cont_b1, 0, nullptr, hcont, 0, BSZ * NCONTN, DM, DM, 1);
  gelu16<<<(BSZ * NCONTN * DM + 255) / 256, 256, 0, stream>>>(
      hcont, (long long)BSZ * NCONTN * DM);
  gemm(hcont, 0, cW2H, 0, cont_b2, 0, ycont2, nullptr, 0, BSZ * NCONTN, 1, DM, 1);
  out_cont<<<2, 256, 0, stream>>>(
      ycont2, out + (long long)BSZ * CIN * SEQL + (long long)BSZ * NCATN * VOC);
}
